// VariableSelectionNetwork_4964982194346
// MI455X (gfx1250) — matrix operand order repaired and checked by reading, not compiled, not run on hardware
//
#include <hip/hip_runtime.h>

#define Bn 32
#define Tn 512
#define Fn 32
#define Dn 256
#define ROWS (Bn * Tn)            // 16384 (b,t) rows
#define WN (Fn * Dn * Dn)         // 2097152 elements per big weight tensor
#define LDSH 264                  // h-tile row stride (ushorts), padded vs 256
#define LDSW 40                   // transposed weight-slice col stride (ushorts), padded vs 32
#define SLICE 8192                // ushorts per packed 32x256 weight K-slice

#define VSN_USE_ASYNC 1           // global_load_async_to_lds_b128 staging path

typedef __attribute__((ext_vector_type(16))) __bf16 v16bf;
typedef __attribute__((ext_vector_type(8)))  float  v8f;
typedef __attribute__((ext_vector_type(8)))  unsigned short us8;

__device__ __forceinline__ unsigned short f32_bf16(float f) {
  unsigned int u = __builtin_bit_cast(unsigned int, f);
  u += 0x7FFFu + ((u >> 16) & 1u);              // round-to-nearest-even
  return (unsigned short)(u >> 16);
}
__device__ __forceinline__ __bf16 us_bf(unsigned short s) {
  return __builtin_bit_cast(__bf16, s);
}
__device__ __forceinline__ float sigmoidf_(float v) {
  return 1.0f / (1.0f + __expf(-v));
}

// Async DMA one 16B chunk: global -> LDS (tracked by ASYNCcnt).
__device__ __forceinline__ void async_copy_b128(unsigned lds_byte_off,
                                                const void* gaddr) {
  asm volatile("global_load_async_to_lds_b128 %0, %1, off"
               :: "v"(lds_byte_off), "v"(gaddr) : "memory");
}
__device__ __forceinline__ void wait_async0() {
#if __has_builtin(__builtin_amdgcn_s_wait_asynccnt)
  __builtin_amdgcn_s_wait_asynccnt(0);
#else
  asm volatile("s_wait_asynccnt 0x0" ::: "memory");
#endif
}

// Build a v16bf fragment from two aligned 8x16-bit LDS chunks.
__device__ __forceinline__ v16bf load_frag16(const unsigned short* p0,
                                             const unsigned short* p1) {
  us8 lo = *(const us8*)p0;
  us8 hi = *(const us8*)p1;
  v16bf r;
#pragma unroll
  for (int i = 0; i < 8; ++i) {
    r[i]     = us_bf(lo[i]);
    r[8 + i] = us_bf(hi[i]);
  }
  return r;
}

// Stage one packed K-slice (SLICE ushorts, [n][k] k-minor) into a padded
// [256][LDSW] LDS buffer. 1024 16B chunks over 256 threads. Async: DMA in
// flight when this returns; consumer must wait_async0() + barrier.
__device__ __forceinline__ void stage_slice(unsigned short* dst,
                                            const unsigned short* src,
                                            int tid) {
#pragma unroll
  for (int it = 0; it < 4; ++it) {
    int c = tid + it * 256;                       // chunk id 0..1023
    int loff = (c >> 2) * LDSW + (c & 3) * 8;     // n*LDSW + k0
#if VSN_USE_ASYNC
    async_copy_b128((unsigned)(size_t)&dst[loff], src + c * 8);
#else
    *(us8*)&dst[loff] = *(const us8*)(src + c * 8);
#endif
  }
  __builtin_prefetch(src + SLICE, 0, 3);          // pull slice-after-next toward L2
}

// One K-step: A fragment from h-tile (abuf, row-major K) x 4 B fragments
// from a staged weight slice (bbuf, [n][k]) -> 4 WMMA accumulators.
__device__ __forceinline__ void gemm_step(const unsigned short* abuf, int ks,
                                          const unsigned short* bbuf,
                                          int mtile, int nquad, int l15, int hsel,
                                          v8f acc[4]) {
  const unsigned short* pa =
      &abuf[(mtile * 16 + l15) * LDSH + ks * 32 + hsel * 8];
  v16bf afr = load_frag16(pa, pa + 16);
#pragma unroll
  for (int j = 0; j < 4; ++j) {
    const unsigned short* pbm =
        &bbuf[((nquad * 4 + j) * 16 + l15) * LDSW + hsel * 8];
    v16bf bfr = load_frag16(pbm, pbm + 16);
    acc[j] = __builtin_amdgcn_wmma_f32_16x16x32_bf16(
        false, afr, false, bfr, (short)0, acc[j], false, false);
  }
}

// ---------------- kernel 1: fp32 -> bf16 weight conversion + packing ----------------
// Destination layout: [t][f][ks][n][k]  (t in {w2,wp,wg}; k minor, 32 wide)
__global__ void vsn_cvt_kernel(const float* __restrict__ w2,
                               const float* __restrict__ wp,
                               const float* __restrict__ wg,
                               unsigned short* __restrict__ o) {
  int i = blockIdx.x * blockDim.x + threadIdx.x;  // source-linear (coalesced reads)
  if (i < WN) {
    int n  = i & 255;
    int kk = (i >> 8) & 255;
    int f  = i >> 16;
    int ks = kk >> 5;
    int k  = kk & 31;
    size_t dst = ((size_t)f * 8 + ks) * SLICE + n * 32 + k;
    o[dst]                           = f32_bf16(w2[i]);
    o[(size_t)8 * Fn * SLICE + dst]  = f32_bf16(wp[i]);
    o[(size_t)16 * Fn * SLICE + dst] = f32_bf16(wg[i]);
  }
}

// ---------------- kernel 2: gate GRN (F->F) + softmax -> weights [ROWS,F] ----------------
__global__ __launch_bounds__(256) void vsn_gate_kernel(
    const float* __restrict__ x,
    const float* __restrict__ gw1, const float* __restrict__ gb1,
    const float* __restrict__ gw2, const float* __restrict__ gb2,
    const float* __restrict__ gwp, const float* __restrict__ gbp,
    const float* __restrict__ gwg, const float* __restrict__ gbg,
    const float* __restrict__ gga, const float* __restrict__ gbe,
    const float* __restrict__ pw,  const float* __restrict__ pb,
    float* __restrict__ wout) {
  const int lane = threadIdx.x & 31;
  const int row  = blockIdx.x * (blockDim.x >> 5) + (threadIdx.x >> 5);
  const float xv = x[row * Fn + lane];

  float a = 0.f;
#pragma unroll
  for (int k = 0; k < Fn; ++k) a += __shfl(xv, k, 32) * gw1[k * Fn + lane];
  a += gb1[lane];
  a = a > 0.f ? a : (__expf(a) - 1.f);          // ELU

  float h = 0.f;
#pragma unroll
  for (int k = 0; k < Fn; ++k) h += __shfl(a, k, 32) * gw2[k * Fn + lane];
  h += gb2[lane];

  float p = 0.f, g = 0.f;
#pragma unroll
  for (int k = 0; k < Fn; ++k) {
    float hv = __shfl(h, k, 32);
    p += hv * gwp[k * Fn + lane];
    g += hv * gwg[k * Fn + lane];
  }
  p += gbp[lane];
  g += gbg[lane];
  float t = p * sigmoidf_(g) + xv;              // GLU + identity skip

  float s = t;
#pragma unroll
  for (int m = 16; m > 0; m >>= 1) s += __shfl_xor(s, m, 32);
  const float mu = s * (1.f / 32.f);
  float d = t - mu;
  float q = d * d;
#pragma unroll
  for (int m = 16; m > 0; m >>= 1) q += __shfl_xor(q, m, 32);
  const float var = q * (1.f / 32.f);
  const float wn = d * rsqrtf(var + 1e-5f) * gga[lane] + gbe[lane];

  float lg = 0.f;
#pragma unroll
  for (int k = 0; k < Fn; ++k) lg += __shfl(wn, k, 32) * pw[k * Fn + lane];
  lg += pb[lane];

  float mx = lg;
#pragma unroll
  for (int m = 16; m > 0; m >>= 1) mx = fmaxf(mx, __shfl_xor(mx, m, 32));
  float e = __expf(lg - mx);
  float se = e;
#pragma unroll
  for (int m = 16; m > 0; m >>= 1) se += __shfl_xor(se, m, 32);
  wout[row * Fn + lane] = e / se;
}

// ---------------- kernel 3: fused per-feature GRNs + weighted selection ----------------
// Double-buffered async weight staging: slice ks+1 DMA overlaps WMMAs on slice ks.
__global__ __launch_bounds__(256) void vsn_main_kernel(
    const float* __restrict__ x,
    const float* __restrict__ fw1, const float* __restrict__ fb1,
    const unsigned short* __restrict__ wbf,      // packed bf16 [w2 | wp | wg]
    const float* __restrict__ fb2,
    const float* __restrict__ fbp, const float* __restrict__ fbg,
    const float* __restrict__ fwsk, const float* __restrict__ fbsk,
    const float* __restrict__ fga, const float* __restrict__ fbe,
    const float* __restrict__ wts,               // softmax weights [ROWS,F]
    float* __restrict__ outsel) {                // [ROWS,D]
  __shared__ unsigned short ldsA[10240];   // h1 [32][LDSH] ; later wp/wg pong buffer
  __shared__ unsigned short ldsB[10240];   // w2 pong buffer ; later h2 [32][LDSH]
  __shared__ unsigned short ldsW[10240];   // weight slice ping buffer [256][LDSW]
  __shared__ float ldsMu[32];
  __shared__ float ldsRs[32];

  const int tid   = threadIdx.x;
  const int lane  = tid & 31;
  const int hsel  = lane >> 4;        // 0: lanes 0-15, 1: lanes 16-31
  const int l15   = lane & 15;
  const int wid   = tid >> 5;
  const int mtile = wid >> 2;         // 0..1
  const int nquad = wid & 3;          // 0..3
  const int rowbase = blockIdx.x * 32;

  float accSel[4][8];
#pragma unroll
  for (int j = 0; j < 4; ++j)
#pragma unroll
    for (int v = 0; v < 8; ++v) accSel[j][v] = 0.f;

  for (int f = 0; f < Fn; ++f) {
    const unsigned short* w2s = wbf + (size_t)f * 8 * SLICE;
    const unsigned short* wps = wbf + ((size_t)8 * Fn + (size_t)f * 8) * SLICE;
    const unsigned short* wgs = wbf + ((size_t)16 * Fn + (size_t)f * 8) * SLICE;

    if (tid < 32) { ldsMu[tid] = 0.f; ldsRs[tid] = 0.f; }

    // ---- h1 = elu(x * w1[f] + b1[f]) -> ldsA (bf16, b128 stores) ----
    {
      int r   = tid >> 3;             // 32 rows, 8 threads per row
      int dc0 = (tid & 7) * 32;       // 32 contiguous cols per thread
      float xv = x[(rowbase + r) * Fn + f];
      const float* w1p = fw1 + f * Dn + dc0;
      const float* b1p = fb1 + f * Dn + dc0;
#pragma unroll
      for (int c = 0; c < 4; ++c) {
        us8 pk;
#pragma unroll
        for (int q = 0; q < 8; ++q) {
          float v = xv * w1p[c * 8 + q] + b1p[c * 8 + q];
          v = v > 0.f ? v : (__expf(v) - 1.f);
          pk[q] = f32_bf16(v);
        }
        *(us8*)&ldsA[r * LDSH + dc0 + c * 8] = pk;
      }
    }

    // ---- GEMM1: h2 = h1 @ w2[f]  (ping=ldsW, pong=ldsB) ----
    v8f accH[4] = {};
    stage_slice(ldsW, w2s, tid);                  // slice 0 in flight
    for (int ks = 0; ks < 8; ++ks) {
      unsigned short* cur = (ks & 1) ? ldsB : ldsW;
      unsigned short* nxt = (ks & 1) ? ldsW : ldsB;
      wait_async0();                              // current slice landed (per wave)
      __syncthreads();                            // visible to all; old reads done
      if (ks < 7) stage_slice(nxt, w2s + (ks + 1) * SLICE, tid);  // overlap
      gemm_step(ldsA, ks, cur, mtile, nquad, l15, hsel, accH);
    }
    __syncthreads();                              // last-slice reads done (ldsB)

    // ---- h2 + b2 -> ldsB (bf16, WMMA C layout scatter) ----
#pragma unroll
    for (int j = 0; j < 4; ++j) {
      int col = (nquad * 4 + j) * 16 + l15;
      float bb = fb2[f * Dn + col];
#pragma unroll
      for (int v = 0; v < 8; ++v) {
        int r = mtile * 16 + v + 8 * hsel;
        ldsB[r * LDSH + col] = f32_bf16(accH[j][v] + bb);
      }
    }

    // ---- GEMM2: p = h2 @ wp[f]  (ping=ldsW, pong=ldsA; h1 dead) ----
    v8f accP[4] = {};
    stage_slice(ldsW, wps, tid);
    for (int ks = 0; ks < 8; ++ks) {
      unsigned short* cur = (ks & 1) ? ldsA : ldsW;
      unsigned short* nxt = (ks & 1) ? ldsW : ldsA;
      wait_async0();
      __syncthreads();                            // also orders h2 stores at ks==0
      if (ks < 7) stage_slice(nxt, wps + (ks + 1) * SLICE, tid);
      gemm_step(ldsB, ks, cur, mtile, nquad, l15, hsel, accP);
    }
    __syncthreads();

    // ---- GEMM3: g = h2 @ wg[f]  (ping=ldsW, pong=ldsA) ----
    v8f accG[4] = {};
    stage_slice(ldsW, wgs, tid);
    for (int ks = 0; ks < 8; ++ks) {
      unsigned short* cur = (ks & 1) ? ldsA : ldsW;
      unsigned short* nxt = (ks & 1) ? ldsW : ldsA;
      wait_async0();
      __syncthreads();
      if (ks < 7) stage_slice(nxt, wgs + (ks + 1) * SLICE, tid);
      gemm_step(ldsB, ks, cur, mtile, nquad, l15, hsel, accG);
    }

    // ---- GLU + skip, LayerNorm stats (half-wave shuffles + LDS atomics) ----
    float xr[8], wtr[8];
#pragma unroll
    for (int v = 0; v < 8; ++v) {
      int gr = rowbase + mtile * 16 + v + 8 * hsel;
      xr[v]  = x[gr * Fn + f];
      wtr[v] = wts[gr * Fn + f];
    }
    float sreg[4][8];
    float rsum[8], rsq[8];
#pragma unroll
    for (int v = 0; v < 8; ++v) { rsum[v] = 0.f; rsq[v] = 0.f; }
#pragma unroll
    for (int j = 0; j < 4; ++j) {
      int col = (nquad * 4 + j) * 16 + l15;
      float bp  = fbp[f * Dn + col],  bg  = fbg[f * Dn + col];
      float wsv = fwsk[f * Dn + col], bsv = fbsk[f * Dn + col];
#pragma unroll
      for (int v = 0; v < 8; ++v) {
        float pv = accP[j][v] + bp;
        float gv = accG[j][v] + bg;
        float sv = pv * sigmoidf_(gv) + xr[v] * wsv + bsv;  // GLU + skip(1->D)
        sreg[j][v] = sv;
        rsum[v] += sv;
        rsq[v]  += sv * sv;
      }
    }
#pragma unroll
    for (int v = 0; v < 8; ++v) {
      float ssum = rsum[v], qsum = rsq[v];
#pragma unroll
      for (int m = 1; m < 16; m <<= 1) {
        ssum += __shfl_xor(ssum, m, 16);
        qsum += __shfl_xor(qsum, m, 16);
      }
      if (l15 == 0) {
        int r = mtile * 16 + v + 8 * hsel;
        atomicAdd(&ldsMu[r], ssum);
        atomicAdd(&ldsRs[r], qsum);
      }
    }
    __syncthreads();
    if (tid < 32) {
      float mu  = ldsMu[tid] * (1.f / 256.f);
      float var = ldsRs[tid] * (1.f / 256.f) - mu * mu;
      ldsMu[tid] = mu;
      ldsRs[tid] = rsqrtf(var + 1e-5f);
    }
    __syncthreads();

    // ---- normalize + weighted accumulate into persistent selected regs ----
#pragma unroll
    for (int j = 0; j < 4; ++j) {
      int col = (nquad * 4 + j) * 16 + l15;
      float ga = fga[f * Dn + col], be = fbe[f * Dn + col];
#pragma unroll
      for (int v = 0; v < 8; ++v) {
        int r = mtile * 16 + v + 8 * hsel;
        float sn = (sreg[j][v] - ldsMu[r]) * ldsRs[r] * ga + be;
        accSel[j][v] += sn * wtr[v];
      }
    }
    __syncthreads();   // protects ldsMu/ldsRs re-zero + ldsA/ldsB reuse next f
  }

  // ---- write selected [ROWS,D] ----
#pragma unroll
  for (int j = 0; j < 4; ++j) {
    int col = (nquad * 4 + j) * 16 + l15;
#pragma unroll
    for (int v = 0; v < 8; ++v) {
      int gr = rowbase + mtile * 16 + v + 8 * hsel;
      outsel[(size_t)gr * Dn + col] = accSel[j][v];
    }
  }
}

extern "C" void kernel_launch(void* const* d_in, const int* in_sizes, int n_in,
                              void* d_out, int out_size, void* d_ws, size_t ws_size,
                              hipStream_t stream) {
  (void)in_sizes; (void)n_in; (void)out_size; (void)ws_size;
  const float* x    = (const float*)d_in[0];
  const float* fw1  = (const float*)d_in[1];
  const float* fb1  = (const float*)d_in[2];
  const float* fw2  = (const float*)d_in[3];
  const float* fb2  = (const float*)d_in[4];
  const float* fwp  = (const float*)d_in[5];
  const float* fbp  = (const float*)d_in[6];
  const float* fwg  = (const float*)d_in[7];
  const float* fbg  = (const float*)d_in[8];
  const float* fws_ = (const float*)d_in[9];
  const float* fbs_ = (const float*)d_in[10];
  const float* fga  = (const float*)d_in[11];
  const float* fbe  = (const float*)d_in[12];
  const float* gw1  = (const float*)d_in[13];
  const float* gb1  = (const float*)d_in[14];
  const float* gw2  = (const float*)d_in[15];
  const float* gb2  = (const float*)d_in[16];
  const float* gwp  = (const float*)d_in[17];
  const float* gbp  = (const float*)d_in[18];
  const float* gwg  = (const float*)d_in[19];
  const float* gbg  = (const float*)d_in[20];
  const float* gga  = (const float*)d_in[21];
  const float* gbe  = (const float*)d_in[22];
  const float* pw   = (const float*)d_in[23];
  const float* pb   = (const float*)d_in[24];

  float* outsel = (float*)d_out;
  float* outw   = outsel + (size_t)ROWS * Dn;
  unsigned short* wbf = (unsigned short*)d_ws;   // 12 MB packed bf16 weights

  vsn_cvt_kernel<<<(WN + 255) / 256, 256, 0, stream>>>(fw2, fwp, fwg, wbf);
  vsn_gate_kernel<<<ROWS / 8, 256, 0, stream>>>(x, gw1, gb1, gw2, gb2, gwp, gbp,
                                                gwg, gbg, gga, gbe, pw, pb, outw);
  vsn_main_kernel<<<ROWS / 32, 256, 0, stream>>>(x, fw1, fb1, wbf, fb2, fbp, fbg,
                                                 fws_, fbs_, fga, fbe, outw, outsel);
}
